// RelativeGlobalAttention_20968030339750
// MI455X (gfx1250) — matrix operand order repaired and checked by reading, not compiled, not run on hardware
//
#include <hip/hip_runtime.h>
#include <hip/hip_bf16.h>

typedef __bf16 bf16_t;
typedef __attribute__((ext_vector_type(16))) __bf16 v16bf;
typedef __attribute__((ext_vector_type(8)))  __bf16 v8bf;
typedef __attribute__((ext_vector_type(4)))  __bf16 v4bf;
typedef __attribute__((ext_vector_type(8)))  float  v8f;

#define S_LEN  2048
#define NH     8
#define NB     2
#define DMODEL 512
#define DHEAD  64
#define MROWS  (NB * S_LEN)   // 4096
#define NEG_BIG (-1.0e9f)

#define V8F_ZERO ((v8f){0.f,0.f,0.f,0.f,0.f,0.f,0.f,0.f})

__device__ __forceinline__ v8f wmma_bf16(v16bf a, v16bf b, v8f c) {
  // D = A(16x32 bf16) * B(32x16 bf16) + C(16x16 f32)
  return __builtin_amdgcn_wmma_f32_16x16x32_bf16(false, a, false, b, (short)0, c,
                                                 false, false);
}

// Async global->LDS 16-byte copy (CDNA5 GLOBAL_LOAD_ASYNC_TO_LDS_B128,
// tracked by ASYNCcnt). lds_addr = wave-LDS byte offset (= low 32 bits of the
// flat pointer for the LDS aperture), gaddr = 16B-aligned global address.
__device__ __forceinline__ void async_load_b128(unsigned lds_addr,
                                                const void* gaddr) {
  asm volatile("global_load_async_to_lds_b128 %0, %1, off"
               :: "v"(lds_addr), "v"(gaddr) : "memory");
}
__device__ __forceinline__ void wait_async0() {
  asm volatile("s_wait_asynccnt 0x0" ::: "memory");
}
__device__ __forceinline__ unsigned lds_off(const void* p) {
  return (unsigned)(unsigned long long)p;  // ISA: LDS aperture addr[31:0]
}

// A-matrix fragment from row-major LDS tile (MxK), per CDNA5 16-bit A layout:
// lane<16: row=lane, K 0..7 in v[0..3], K16..23 in v[4..7]
// lane>=16: row=lane-16, K 8..15 / K 24..31
__device__ __forceinline__ v16bf load_a_frag(const bf16_t* p0, int lda,
                                             int row0, int k0, int lane) {
  const int r = lane & 15, half = lane >> 4;
  const bf16_t* p = p0 + (row0 + r) * lda + k0 + half * 8;
  v8bf lo = *(const v8bf*)p;        // K 0..7  (+8 if upper half)
  v8bf hi = *(const v8bf*)(p + 16); // K 16..23 (+8 if upper half)
  v16bf o;
#pragma unroll
  for (int i = 0; i < 8; ++i) { o[i] = lo[i]; o[i + 8] = hi[i]; }
  return o;
}

// B-matrix fragment from LDS tile stored [N][K] row-major, per CDNA5 layout:
// lane<16: col N=lane, K 0..7 / K 16..23; lane>=16: col N=lane-16, K 8..15 / K 24..31
__device__ __forceinline__ v16bf load_b_frag(const bf16_t* p0, int ldb,
                                             int n0, int k0, int lane) {
  const int n = lane & 15, half = lane >> 4;
  const bf16_t* p = p0 + (n0 + n) * ldb + k0 + half * 8;
  v8bf lo = *(const v8bf*)p;
  v8bf hi = *(const v8bf*)(p + 16);
  v16bf o;
#pragma unroll
  for (int i = 0; i < 8; ++i) { o[i] = lo[i]; o[i + 8] = hi[i]; }
  return o;
}

// ---------------------------------------------------------------------------
// GEMM: C[64x64 tile] = A[M x 512] * W[512 x 512] + bias
// MODE 0: A fp32, result -> bf16 head-major [B,H,S,DH]  (QKV projections)
// MODE 1: A bf16 (ctx),  result -> fp32 row-major [M,512] (output projection)
// ---------------------------------------------------------------------------
template <int MODE>
__global__ void __launch_bounds__(128)
rga_gemm512(const float* __restrict__ Af32, const bf16_t* __restrict__ Abf,
            const float* __restrict__ W, const float* __restrict__ bias,
            bf16_t* __restrict__ dst_bf, float* __restrict__ dst_f32) {
  __shared__ bf16_t As[64][40];
  __shared__ bf16_t Bs[64][40];  // W transposed: [n][k]
  const int t = threadIdx.x, lane = t & 31, w = t >> 5;
  const int m0 = blockIdx.y * 64, n0 = blockIdx.x * 64;

  v8f acc[4];
#pragma unroll
  for (int i = 0; i < 4; ++i) acc[i] = V8F_ZERO;

  for (int kk = 0; kk < DMODEL; kk += 32) {
    // A tile: 64 rows x 32 K
    if (MODE == 0) {
#pragma unroll
      for (int q = 0; q < 4; ++q) {
        int id = q * 128 + t;
        int row = id >> 3, c4 = (id & 7) * 4;
        float4 v = *(const float4*)(Af32 + (size_t)(m0 + row) * DMODEL + kk + c4);
        As[row][c4 + 0] = (bf16_t)v.x; As[row][c4 + 1] = (bf16_t)v.y;
        As[row][c4 + 2] = (bf16_t)v.z; As[row][c4 + 3] = (bf16_t)v.w;
      }
    } else {
      // bf16 -> bf16: pure copy, use the async global->LDS engine
#pragma unroll
      for (int q = 0; q < 4; ++q) {
        int id = q * 128 + t;
        int row = id >> 2, c8 = (id & 3) * 8;  // 16B chunks, 4 per row
        async_load_b128(lds_off(&As[row][c8]),
                        Abf + (size_t)(m0 + row) * DMODEL + kk + c8);
      }
    }
    // W tile: 32 K x 64 N, transposed into Bs[n][k] (f32 -> bf16 convert)
#pragma unroll
    for (int q = 0; q < 4; ++q) {
      int id = q * 128 + t;
      int kr = id >> 4, c4 = (id & 15) * 4;
      float4 v = *(const float4*)(W + (size_t)(kk + kr) * DMODEL + n0 + c4);
      Bs[c4 + 0][kr] = (bf16_t)v.x; Bs[c4 + 1][kr] = (bf16_t)v.y;
      Bs[c4 + 2][kr] = (bf16_t)v.z; Bs[c4 + 3][kr] = (bf16_t)v.w;
    }
    if (MODE == 1) wait_async0();
    __syncthreads();
    v16bf a = load_a_frag(&As[0][0], 40, w * 16, 0, lane);
#pragma unroll
    for (int ni = 0; ni < 4; ++ni) {
      v16bf b = load_b_frag(&Bs[0][0], 40, ni * 16, 0, lane);
      acc[ni] = wmma_bf16(a, b, acc[ni]);
    }
    __syncthreads();
  }

  const int half = lane >> 4;
#pragma unroll
  for (int ni = 0; ni < 4; ++ni) {
#pragma unroll
    for (int r = 0; r < 8; ++r) {
      int gm = m0 + w * 16 + half * 8 + r;
      int gn = n0 + ni * 16 + (lane & 15);
      float v = acc[ni][r] + bias[gn];
      if (MODE == 0) {
        int b = gm >> 11, s = gm & (S_LEN - 1);
        int h = gn >> 6, d = gn & 63;
        dst_bf[(((size_t)(b * NH + h)) * S_LEN + s) * DHEAD + d] = (bf16_t)v;
      } else {
        dst_f32[(size_t)gm * DMODEL + gn] = v;
      }
    }
  }
}

// ---------------------------------------------------------------------------
// Logits: per (bh, 64x64 tile): (Q K^T + srel)/8 with causal mask.
// srel[l,j] = q[l] . E[2047-(l-j)] for j<=l (Toeplitz gather of T = Q @ E_win^T)
// Tiles strictly above the diagonal are filled with -1e9 (softmax-exact zeros).
// ---------------------------------------------------------------------------
__global__ void __launch_bounds__(128)
rga_logits(const bf16_t* __restrict__ qws, const bf16_t* __restrict__ kws,
           const float* __restrict__ E, float* __restrict__ attn) {
  const int jt = blockIdx.x, lt = blockIdx.y, bh = blockIdx.z;
  const int t = threadIdx.x, lane = t & 31, w = t >> 5;
  const int l0 = lt * 64, j0 = jt * 64;
  float* arow = attn + (size_t)bh * S_LEN * S_LEN;

  if (jt > lt) {  // fully masked tile
    const float4 mfill = make_float4(NEG_BIG, NEG_BIG, NEG_BIG, NEG_BIG);
#pragma unroll
    for (int q = 0; q < 8; ++q) {
      int id = q * 128 + t;
      int row = id >> 4, c4 = (id & 15) * 4;
      *(float4*)(arow + (size_t)(l0 + row) * S_LEN + j0 + c4) = mfill;
    }
    return;
  }

  __shared__ bf16_t Qs[64][72];
  __shared__ bf16_t Ks[64][72];
  __shared__ bf16_t Es[128][72];   // E window rows e0..e0+127 (bf16)
  __shared__ float  Ts[64][132];   // T[i][m] = Q[i] . E[e0+m]

  // Q / K tiles: bf16 workspace -> bf16 LDS, pure copy -> async engine.
  // 16B chunks: 8 per 64-elem row; 512 chunks each over 128 threads.
#pragma unroll
  for (int q = 0; q < 4; ++q) {
    int id = q * 128 + t;
    int row = id >> 3, c8 = (id & 7) * 8;
    async_load_b128(lds_off(&Qs[row][c8]),
                    qws + ((size_t)bh * S_LEN + l0 + row) * DHEAD + c8);
    async_load_b128(lds_off(&Ks[row][c8]),
                    kws + ((size_t)bh * S_LEN + j0 + row) * DHEAD + c8);
  }
  // E window: e0 = 1984 - l0 + j0  (>= 0 for j0 <= l0); OOB rows -> 0.
  // f32 -> bf16 conversion, so this one stays on the VALU path.
  const int e0 = 1984 - l0 + j0;
#pragma unroll
  for (int q = 0; q < 16; ++q) {
    int id = q * 128 + t;
    int row = id >> 4, c4 = (id & 15) * 4;
    int er = e0 + row;
    float4 v = (er >= 0 && er < S_LEN)
                   ? *(const float4*)(E + (size_t)er * DHEAD + c4)
                   : make_float4(0.f, 0.f, 0.f, 0.f);
    Es[row][c4 + 0] = (bf16_t)v.x; Es[row][c4 + 1] = (bf16_t)v.y;
    Es[row][c4 + 2] = (bf16_t)v.z; Es[row][c4 + 3] = (bf16_t)v.w;
  }
  wait_async0();
  __syncthreads();

  v16bf a0 = load_a_frag(&Qs[0][0], 72, w * 16, 0, lane);
  v16bf a1 = load_a_frag(&Qs[0][0], 72, w * 16, 32, lane);
  const int half = lane >> 4;

  // Pass 1: T = Q_tile(64x64) @ E_win^T(64x128)
#pragma unroll
  for (int ni = 0; ni < 8; ++ni) {
    v8f acc = V8F_ZERO;
    acc = wmma_bf16(a0, load_b_frag(&Es[0][0], 72, ni * 16, 0, lane), acc);
    acc = wmma_bf16(a1, load_b_frag(&Es[0][0], 72, ni * 16, 32, lane), acc);
#pragma unroll
    for (int r = 0; r < 8; ++r)
      Ts[w * 16 + half * 8 + r][ni * 16 + (lane & 15)] = acc[r];
  }
  __syncthreads();

  // Pass 2: QK^T + skewed bias gather + mask, write fp32 logits
#pragma unroll
  for (int ni = 0; ni < 4; ++ni) {
    v8f acc = V8F_ZERO;
    acc = wmma_bf16(a0, load_b_frag(&Ks[0][0], 72, ni * 16, 0, lane), acc);
    acc = wmma_bf16(a1, load_b_frag(&Ks[0][0], 72, ni * 16, 32, lane), acc);
#pragma unroll
    for (int r = 0; r < 8; ++r) {
      int i = w * 16 + half * 8 + r;
      int jj = ni * 16 + (lane & 15);
      int l = l0 + i, j = j0 + jj;
      float v = (acc[r] + Ts[i][63 + jj - i]) * 0.125f;  // /sqrt(DH)
      if (j > l) v = NEG_BIG;
      arow[(size_t)l * S_LEN + j] = v;
    }
  }
}

// ---------------------------------------------------------------------------
// Row softmax in place: one 256-thread block per (bh, l) row of 2048.
// ---------------------------------------------------------------------------
__global__ void __launch_bounds__(256)
rga_softmax(float* __restrict__ attn) {
  const int row = blockIdx.x;
  float* p = attn + (size_t)row * S_LEN;
  const int t = threadIdx.x, lane = t & 31, wid = t >> 5;
  __shared__ float red[8];

  float v[8];
  float m = -3.4e38f;
#pragma unroll
  for (int i = 0; i < 8; ++i) { v[i] = p[t + 256 * i]; m = fmaxf(m, v[i]); }
#pragma unroll
  for (int o = 16; o > 0; o >>= 1) m = fmaxf(m, __shfl_xor(m, o, 32));
  if (lane == 0) red[wid] = m;
  __syncthreads();
  m = red[0];
#pragma unroll
  for (int i = 1; i < 8; ++i) m = fmaxf(m, red[i]);

  float s = 0.f;
#pragma unroll
  for (int i = 0; i < 8; ++i) { v[i] = __expf(v[i] - m); s += v[i]; }
#pragma unroll
  for (int o = 16; o > 0; o >>= 1) s += __shfl_xor(s, o, 32);
  __syncthreads();
  if (lane == 0) red[wid] = s;
  __syncthreads();
  s = 0.f;
#pragma unroll
  for (int i = 0; i < 8; ++i) s += red[i];

  float inv = 1.0f / s;
#pragma unroll
  for (int i = 0; i < 8; ++i) p[t + 256 * i] = v[i] * inv;
}

// ---------------------------------------------------------------------------
// AV: ctx[bh, l, :] = sum_j attn[l,j] * V[j,:]; K-loop truncated at j<=l0+63.
// ctx stored bf16 in [B, S, D] layout for the output projection.
// ---------------------------------------------------------------------------
__global__ void __launch_bounds__(128)
rga_av(const float* __restrict__ attn, const bf16_t* __restrict__ vws,
       bf16_t* __restrict__ cws) {
  __shared__ bf16_t As[64][40];
  __shared__ bf16_t Bs[64][40];  // V transposed: [d][k]
  const int mt = blockIdx.x, bh = blockIdx.y;
  const int t = threadIdx.x, lane = t & 31, w = t >> 5;
  const int l0 = mt * 64;
  const float* ab = attn + (size_t)bh * S_LEN * S_LEN;

  v8f acc[4];
#pragma unroll
  for (int i = 0; i < 4; ++i) acc[i] = V8F_ZERO;

  for (int j0 = 0; j0 < l0 + 64; j0 += 32) {  // causal truncation
#pragma unroll
    for (int q = 0; q < 4; ++q) {
      int id = q * 128 + t;
      int row = id >> 3, c4 = (id & 7) * 4;
      float4 v = *(const float4*)(ab + (size_t)(l0 + row) * S_LEN + j0 + c4);
      As[row][c4 + 0] = (bf16_t)v.x; As[row][c4 + 1] = (bf16_t)v.y;
      As[row][c4 + 2] = (bf16_t)v.z; As[row][c4 + 3] = (bf16_t)v.w;
    }
#pragma unroll
    for (int q = 0; q < 4; ++q) {
      int id = q * 128 + t;
      int vr = id >> 4, c4 = (id & 15) * 4;
      v4bf v = *(const v4bf*)(vws + ((size_t)bh * S_LEN + j0 + vr) * DHEAD + c4);
      Bs[c4 + 0][vr] = v[0]; Bs[c4 + 1][vr] = v[1];
      Bs[c4 + 2][vr] = v[2]; Bs[c4 + 3][vr] = v[3];
    }
    __syncthreads();
    v16bf a = load_a_frag(&As[0][0], 40, w * 16, 0, lane);
#pragma unroll
    for (int ni = 0; ni < 4; ++ni) {
      v16bf b = load_b_frag(&Bs[0][0], 40, ni * 16, 0, lane);
      acc[ni] = wmma_bf16(a, b, acc[ni]);
    }
    __syncthreads();
  }

  const int half = lane >> 4;
  const int b = bh >> 3, h = bh & 7;
#pragma unroll
  for (int ni = 0; ni < 4; ++ni) {
#pragma unroll
    for (int r = 0; r < 8; ++r) {
      int l = l0 + w * 16 + half * 8 + r;
      int d = ni * 16 + (lane & 15);
      cws[((size_t)b * S_LEN + l) * DMODEL + h * DHEAD + d] = (bf16_t)acc[ni][r];
    }
  }
}

// ---------------------------------------------------------------------------
extern "C" void kernel_launch(void* const* d_in, const int* in_sizes, int n_in,
                              void* d_out, int out_size, void* d_ws,
                              size_t ws_size, hipStream_t stream) {
  const float* q_in = (const float*)d_in[0];
  const float* k_in = (const float*)d_in[1];
  const float* v_in = (const float*)d_in[2];
  // d_in[3] = mask (triu, k=1) — applied analytically (causal) in rga_logits
  const float* Wq = (const float*)d_in[4];
  const float* bq = (const float*)d_in[5];
  const float* Wk = (const float*)d_in[6];
  const float* bk = (const float*)d_in[7];
  const float* Wv = (const float*)d_in[8];
  const float* bv = (const float*)d_in[9];
  const float* Wo = (const float*)d_in[10];
  const float* bo = (const float*)d_in[11];
  const float* E  = (const float*)d_in[12];

  // workspace: q,k,v head-major bf16 [16][2048][64] + ctx bf16 [4096][512]
  bf16_t* qws = (bf16_t*)d_ws;
  bf16_t* kws = qws + (size_t)16 * S_LEN * DHEAD;
  bf16_t* vws = kws + (size_t)16 * S_LEN * DHEAD;
  bf16_t* cws = vws + (size_t)16 * S_LEN * DHEAD;

  float* outp  = (float*)d_out;
  float* attnp = outp + (size_t)MROWS * DMODEL;

  dim3 g1(DMODEL / 64, MROWS / 64);  // 8 x 64 tiles
  rga_gemm512<0><<<g1, 128, 0, stream>>>(q_in, nullptr, Wq, bq, qws, nullptr);
  rga_gemm512<0><<<g1, 128, 0, stream>>>(k_in, nullptr, Wk, bk, kws, nullptr);
  rga_gemm512<0><<<g1, 128, 0, stream>>>(v_in, nullptr, Wv, bv, vws, nullptr);

  rga_logits<<<dim3(32, 32, 16), 128, 0, stream>>>(qws, kws, E, attnp);
  rga_softmax<<<dim3(16 * S_LEN), 256, 0, stream>>>(attnp);
  rga_av<<<dim3(32, 16), 128, 0, stream>>>(attnp, vws, cws);
  rga_gemm512<1><<<g1, 128, 0, stream>>>(nullptr, cws, Wo, bo, nullptr, outp);
}
